// TBConv2d_73899207295518
// MI455X (gfx1250) — compile-verified, matrix operand order checked by reading, not run on hardware
//
#include <hip/hip_runtime.h>
#include <hip/hip_bf16.h>

typedef __attribute__((ext_vector_type(16))) _Float16 v16h;
typedef __attribute__((ext_vector_type(8)))  _Float16 v8h;
typedef __attribute__((ext_vector_type(8)))  float    v8f;
typedef __attribute__((ext_vector_type(4)))  int      v4i;

#define C_IN   256
#define C_OUT  256
#define NB     32
#define H      56
#define W      56
#define HW     (H * W)          // 3136
#define NHW    (NB * HW)        // 100352 pixels
#define HP     58
#define WP     58
#define PPIX   (NB * HP * WP)   // 107648 padded pixels

#define LDS_STRIDE 40           // f16 elements per tile row (32 data + 8 pad)

#if __has_builtin(__builtin_amdgcn_global_load_async_to_lds_b128)
#define HAVE_ASYNC_LDS 1
#endif

typedef __attribute__((address_space(1))) v4i* g_v4i_ptr;
typedef __attribute__((address_space(3))) v4i* l_v4i_ptr;

__device__ __forceinline__ void async_copy_b128(const void* g, void* l) {
#ifdef HAVE_ASYNC_LDS
  __builtin_amdgcn_global_load_async_to_lds_b128((g_v4i_ptr)g, (l_v4i_ptr)l, 0, 0);
#else
  *(v8h*)l = *(const v8h*)g;  // synchronous fallback
#endif
}

// ---------------------------------------------------------------------------
// Kernel 1: per-channel batch statistics -> affine scale/shift
// ---------------------------------------------------------------------------
__global__ __launch_bounds__(256) void bn_stats_kernel(
    const float* __restrict__ x, const float* __restrict__ gamma,
    const float* __restrict__ beta, float* __restrict__ sc, float* __restrict__ sh) {
  const int c = blockIdx.x;
  __shared__ float ssum[256];
  __shared__ float ssq[256];
  float s = 0.f, q = 0.f;
  for (int i = threadIdx.x; i < NHW; i += 256) {
    const int n = i / HW;
    const int hw = i - n * HW;
    const float v = x[(size_t)(n * C_IN + c) * HW + hw];
    s += v;
    q += v * v;
  }
  ssum[threadIdx.x] = s;
  ssq[threadIdx.x] = q;
  __syncthreads();
  for (int st = 128; st > 0; st >>= 1) {
    if ((int)threadIdx.x < st) {
      ssum[threadIdx.x] += ssum[threadIdx.x + st];
      ssq[threadIdx.x]  += ssq[threadIdx.x + st];
    }
    __syncthreads();
  }
  if (threadIdx.x == 0) {
    const float mean = ssum[0] * (1.0f / (float)NHW);
    const float var  = ssq[0] * (1.0f / (float)NHW) - mean * mean;
    const float inv  = rsqrtf(var + 1e-4f);
    const float scale = gamma[c] * inv;
    sc[c] = scale;
    sh[c] = beta[c] - mean * scale;
  }
}

// ---------------------------------------------------------------------------
// Kernel 2: normalize + ternarize -> padded NHWC f16 buffer [NB, 58, 58, 256]
// ---------------------------------------------------------------------------
__global__ __launch_bounds__(256) void ternarize_kernel(
    const float* __restrict__ x, const float* __restrict__ sc,
    const float* __restrict__ sh, _Float16* __restrict__ t) {
  const int p = blockIdx.x * 256 + threadIdx.x;  // padded pixel index
  if (p >= PPIX) return;
  const int n  = p / (HP * WP);
  const int r  = p - n * (HP * WP);
  const int hp = r / WP;
  const int wp = r - hp * WP;
  _Float16* dst = t + (size_t)p * C_IN;
  if (hp == 0 || hp == HP - 1 || wp == 0 || wp == WP - 1) {
    v8h z = {};
    #pragma unroll
    for (int c0 = 0; c0 < C_IN; c0 += 8) *(v8h*)(dst + c0) = z;
    return;
  }
  const int h = hp - 1, w = wp - 1;
  const int hw = h * W + w;
  const float* xs = x + (size_t)n * C_IN * HW + hw;
  for (int c0 = 0; c0 < C_IN; c0 += 8) {
    v8h v;
    #pragma unroll
    for (int j = 0; j < 8; ++j) {
      const int c = c0 + j;
      const float xn = xs[(size_t)c * HW] * sc[c] + sh[c];
      v[j] = (xn <= 0.f) ? (_Float16)(-1.0f) : (_Float16)(1.0f);
    }
    *(v8h*)(dst + c0) = v;
  }
}

// ---------------------------------------------------------------------------
// Kernel 3: repack weights OIHW fp32 -> [tap(9)][co(256)][ci(256)] f16
// ---------------------------------------------------------------------------
__global__ __launch_bounds__(256) void repack_w_kernel(
    const float* __restrict__ wsrc, _Float16* __restrict__ w2) {
  const int i = blockIdx.x * 256 + threadIdx.x;
  if (i >= 9 * C_OUT * C_IN) return;
  const int k9 = i >> 16;
  const int rem = i & 65535;
  const int co = rem >> 8;
  const int ci = rem & 255;
  w2[i] = (_Float16)wsrc[(size_t)(co * C_IN + ci) * 9 + k9];
}

// ---------------------------------------------------------------------------
// Kernel 4: implicit-GEMM conv, LDS double-buffered, async global->LDS stage,
// V_WMMA_F32_16X16X32_F16 compute.
// Block tile 128co x 64px, 8 waves, wave tile 32co x 32px (2x2 v8f accums).
// K pipeline: 72 steps of K=32 (9 taps x 8 ci-chunks).
// ---------------------------------------------------------------------------
__global__ __launch_bounds__(256) void conv_wmma_kernel(
    const _Float16* __restrict__ t, const _Float16* __restrict__ w2,
    const float* __restrict__ bias, float* __restrict__ out) {
  __shared__ _Float16 As[2][128 * LDS_STRIDE];
  __shared__ _Float16 Bs[2][64 * LDS_STRIDE];

  const int tid  = threadIdx.x;
  const int lane = tid & 31;
  const int wave = tid >> 5;
  const int m    = lane & 15;
  const int half = lane >> 4;

  const int px_block = blockIdx.x * 64;
  const int co_block = blockIdx.y * 128;
  const int wave_co  = (wave & 3) * 32;   // 4 waves tile co
  const int wave_px  = (wave >> 2) * 32;  // 2 waves tile px

  // ---- staging assignments (16B chunks) ----
  // A tile: 128 rows x 4 chunks = 512 chunks, 2 per thread.
  const int a_row0 = tid >> 2;            // 0..63
  const int a_col  = (tid & 3) * 8;       // f16 offset within row
  const int a_row1 = a_row0 + 64;
  // B tile: 64 rows x 4 chunks = 256 chunks, 1 per thread.
  const int b_row = tid >> 2;             // 0..63
  const int b_col = (tid & 3) * 8;
  // Precompute this thread's B-row global pixel base (padded NHWC).
  size_t b_gbase;
  {
    const int p  = px_block + b_row;
    const int n  = p / HW;
    const int hw = p - n * HW;
    const int h  = hw / W;
    const int w  = hw - h * W;
    b_gbase = (size_t)((n * HP + h) * WP + w) * C_IN + b_col;
  }

  // ---- per-lane fragment pixel bases (for epilogue addressing) ----
  int fp_n[2], fp_hw[2];
  #pragma unroll
  for (int j = 0; j < 2; ++j) {
    const int p = px_block + wave_px + j * 16 + m;
    fp_n[j]  = p / HW;
    fp_hw[j] = p - fp_n[j] * HW;
  }

  // ---- pipeline ----
  // K-step kk in [0,72): k9 = kk>>3 (tap), ci0 = (kk&7)*32
  auto issue_tile = [&](int kk, int buf) {
    const int k9  = kk >> 3;
    const int ci0 = (kk & 7) * 32;
    const int kh  = k9 / 3;
    const int kw  = k9 - kh * 3;
    const size_t toff = (size_t)(kh * WP + kw) * C_IN + ci0;
    const _Float16* wsrc = w2 + (size_t)k9 * (C_OUT * C_IN) + ci0 + a_col;
    async_copy_b128(wsrc + (size_t)(co_block + a_row0) * C_IN,
                    &As[buf][a_row0 * LDS_STRIDE + a_col]);
    async_copy_b128(wsrc + (size_t)(co_block + a_row1) * C_IN,
                    &As[buf][a_row1 * LDS_STRIDE + a_col]);
    async_copy_b128(t + b_gbase + toff,
                    &Bs[buf][b_row * LDS_STRIDE + b_col]);
  };

  v8f acc[2][2] = {};

  issue_tile(0, 0);

  for (int kk = 0; kk < 72; ++kk) {
    const int buf = kk & 1;
    if (kk + 1 < 72) {
      issue_tile(kk + 1, buf ^ 1);
#ifdef HAVE_ASYNC_LDS
      // 3 in-flight ops belong to tile kk+1; tile kk's have completed (in-order).
      __builtin_amdgcn_s_wait_asynccnt(3);
#endif
    } else {
#ifdef HAVE_ASYNC_LDS
      __builtin_amdgcn_s_wait_asynccnt(0);
#endif
    }
    __syncthreads();      // tile kk visible to all waves

    v16h A[2], B[2];
    #pragma unroll
    for (int i = 0; i < 2; ++i) {
      const _Float16* ap = &As[buf][(wave_co + i * 16 + m) * LDS_STRIDE];
      v8h lo = *(const v8h*)(ap + half * 8);
      v8h hi = *(const v8h*)(ap + 16 + half * 8);
      A[i] = __builtin_shufflevector(lo, hi, 0, 1, 2, 3, 4, 5, 6, 7,
                                             8, 9, 10, 11, 12, 13, 14, 15);
    }
    #pragma unroll
    for (int j = 0; j < 2; ++j) {
      const _Float16* bp = &Bs[buf][(wave_px + j * 16 + m) * LDS_STRIDE];
      v8h lo = *(const v8h*)(bp + half * 8);
      v8h hi = *(const v8h*)(bp + 16 + half * 8);
      B[j] = __builtin_shufflevector(lo, hi, 0, 1, 2, 3, 4, 5, 6, 7,
                                             8, 9, 10, 11, 12, 13, 14, 15);
    }
    #pragma unroll
    for (int i = 0; i < 2; ++i) {
      #pragma unroll
      for (int j = 0; j < 2; ++j) {
        acc[i][j] = __builtin_amdgcn_wmma_f32_16x16x32_f16(
            false, A[i], false, B[j], (short)0, acc[i][j], false, false);
      }
    }
    __syncthreads();      // done reading buf before it is refilled (kk+2)
  }

  // ---- epilogue: bias + ReLU, scatter to NCHW fp32 ----
  // C/D layout: VGPR r, lane l -> M = r + 8*(l>=16), N = l & 15.
  #pragma unroll
  for (int j = 0; j < 2; ++j) {
    #pragma unroll
    for (int i = 0; i < 2; ++i) {
      #pragma unroll
      for (int r = 0; r < 8; ++r) {
        const int co = co_block + wave_co + i * 16 + r + 8 * half;
        float v = acc[i][j][r] + bias[co];
        v = fmaxf(v, 0.f);
        out[(size_t)(fp_n[j] * C_OUT + co) * HW + fp_hw[j]] = v;
      }
    }
  }
}

// ---------------------------------------------------------------------------
// Launcher
// ---------------------------------------------------------------------------
extern "C" void kernel_launch(void* const* d_in, const int* in_sizes, int n_in,
                              void* d_out, int out_size, void* d_ws, size_t ws_size,
                              hipStream_t stream) {
  const float* x     = (const float*)d_in[0];
  const float* gamma = (const float*)d_in[1];
  const float* beta  = (const float*)d_in[2];
  const float* wgt   = (const float*)d_in[3];
  const float* bias  = (const float*)d_in[4];
  float* out = (float*)d_out;

  char* ws = (char*)d_ws;
  float*    sc = (float*)ws;
  float*    sh = sc + 256;
  _Float16* w2 = (_Float16*)(ws + 4096);
  _Float16* t  = (_Float16*)(ws + (2u << 20));

  bn_stats_kernel<<<C_IN, 256, 0, stream>>>(x, gamma, beta, sc, sh);
  ternarize_kernel<<<(PPIX + 255) / 256, 256, 0, stream>>>(x, sc, sh, t);
  repack_w_kernel<<<(9 * C_OUT * C_IN) / 256, 256, 0, stream>>>(wgt, w2);

  dim3 grid(NHW / 64, C_OUT / 128);
  conv_wmma_kernel<<<grid, 256, 0, stream>>>(t, w2, bias, out);
}
